// multi_head_attention_49993419325626
// MI455X (gfx1250) — matrix operand order repaired and checked by reading, not compiled, not run on hardware
//
#include <hip/hip_runtime.h>

// ---------------------------------------------------------------------------
// Types for CDNA5 WMMA (wave32): 16x16x32 bf16 -> f32
// ---------------------------------------------------------------------------
typedef __attribute__((ext_vector_type(16))) __bf16        v16bf;
typedef __attribute__((ext_vector_type(8)))  float         v8f;
typedef __attribute__((ext_vector_type(4)))  unsigned int  u32x4;
typedef __attribute__((ext_vector_type(8)))  int           i32x8;
typedef __attribute__((ext_vector_type(4)))  int           i32x4;

struct F16x16 { u32x4 lo, hi; };

__device__ __forceinline__ v16bf make_frag(u32x4 lo, u32x4 hi) {
  F16x16 t{lo, hi};
  return __builtin_bit_cast(v16bf, t);
}

__device__ __forceinline__ unsigned short f2bf(float f) {
  unsigned int u = __builtin_bit_cast(unsigned int, f);
  u += 0x7FFFu + ((u >> 16) & 1u);   // round-to-nearest-even
  return (unsigned short)(u >> 16);
}

__device__ __forceinline__ v8f wmma_bf16(v16bf a, v16bf b, v8f c) {
  return __builtin_amdgcn_wmma_f32_16x16x32_bf16(false, a, false, b, (short)0, c,
                                                 false, false);
}

// ---------------------------------------------------------------------------
// TDM: 2D tile load Global -> LDS (Tensor Data Mover, gfx1250).
// D# layout per CDNA5 ISA.
// All element quantities are in data_size units (2-byte bf16 here).
// This toolchain exposes the 6-arg builtin: (g0, g1, g2, g3, g1b, cpol).
// ---------------------------------------------------------------------------
__device__ __forceinline__ void tdm_load_2d_bf16(
    unsigned int lds_off, const void* gptr,
    unsigned int tile_d0, unsigned int tile_d1,
    unsigned int tensor_d0, unsigned int tensor_d1,
    unsigned int stride_d0) {
  unsigned long long ga = (unsigned long long)(size_t)gptr;
  u32x4 g0;
  g0[0] = 1u;                                             // count=1 (user D#)
  g0[1] = lds_off;                                        // lds_addr [63:32]
  g0[2] = (unsigned int)(ga & 0xFFFFFFFFu);               // global_addr lo
  g0[3] = (unsigned int)((ga >> 32) & 0x01FFFFFFu) |      // global_addr hi
          (2u << 30);                                     // type=2 ("image")
  i32x8 g1;
  g1[0] = (int)(1u << 16);                 // data_size=1 (2 bytes), mask=0
  g1[1] = (int)(tensor_d0 << 16);          // tensor_dim0[15:0] @ [63:48]
  g1[2] = (int)((tensor_d0 >> 16) | (tensor_d1 << 16));   // dim0 hi | dim1 lo
  g1[3] = (int)((tensor_d1 >> 16) | (tile_d0 << 16));     // dim1 hi | tile_dim0
  g1[4] = (int)(tile_d1 & 0xFFFFu);        // tile_dim1; tile_dim2=0
  g1[5] = (int)stride_d0;                  // tensor_dim0_stride[31:0]
  g1[6] = 0;                               // stride0 hi | dim1_stride lo
  g1[7] = 0;                               // dim1_stride hi
  i32x4 z4 = {0, 0, 0, 0};                 // groups 2/3 unused (2D tile)
  i32x8 z8 = {0, 0, 0, 0, 0, 0, 0, 0};
  __builtin_amdgcn_tensor_load_to_lds(g0, g1, z4, z4, z8, 0);
}

// ---------------------------------------------------------------------------
// fp32 -> bf16 bulk conversion (vectorized, n multiple of 4)
// ---------------------------------------------------------------------------
__global__ __launch_bounds__(256) void cvt_f32_bf16(
    const float* __restrict__ in, unsigned short* __restrict__ out, int n4) {
  int i = blockIdx.x * 256 + threadIdx.x;
  if (i < n4) {
    float4 v = ((const float4*)in)[i];
    ushort4 o;
    o.x = f2bf(v.x); o.y = f2bf(v.y); o.z = f2bf(v.z); o.w = f2bf(v.w);
    ((ushort4*)out)[i] = o;
  }
}

// ---------------------------------------------------------------------------
// GEMM: out[m,n] = sum_k A[m,k] * W[n,k] + bias[n]   (torch Linear: x @ W.T)
// A: [M x K] bf16 row-major; W: [N x K] bf16 row-major.
// Workgroup: 256 thr (8 waves), tile 128x128, K-step 32.
// MODE 0: fp32 out [M x N]; MODE 1: bf16 [B,H,S,64]; MODE 2: bf16 [B,H,64,S].
// ---------------------------------------------------------------------------
template <int MODE>
__global__ __launch_bounds__(256) void gemm_bf16(
    const unsigned short* __restrict__ A, const unsigned short* __restrict__ W,
    const float* __restrict__ bias, void* __restrict__ out,
    int M, int N, int K, int Srows) {
  __shared__ unsigned short At[128 * 32];
  __shared__ unsigned short Bt[128 * 32];

  const int tid    = threadIdx.x;
  const int lane   = tid & 31;
  const int w      = tid >> 5;
  const int laneN  = lane & 15;
  const int hiHalf = lane >> 4;              // 0 | 1
  const int wm     = (w >> 1) * 32;          // wave row base in tile
  const int wn     = (w & 1) * 64;           // wave col base in tile
  const int bm     = blockIdx.y * 128;
  const int bn     = blockIdx.x * 128;

  v8f acc[2][4];
#pragma unroll
  for (int i = 0; i < 2; ++i)
#pragma unroll
    for (int j = 0; j < 4; ++j) acc[i][j] = 0.0f;

  const int srow  = tid >> 1;                // 0..127
  const int shalf = (tid & 1) * 16;          // 0 | 16 (bf16 elements)

  for (int kk = 0; kk < K; kk += 32) {
    __syncthreads();
    {
      const u32x4* pa = (const u32x4*)(A + (size_t)(bm + srow) * K + kk + shalf);
      const u32x4* pb = (const u32x4*)(W + (size_t)(bn + srow) * K + kk + shalf);
      u32x4 a0 = pa[0], a1 = pa[1];
      u32x4 b0 = pb[0], b1 = pb[1];
      if (kk + 32 < K) {                     // gfx1250 global_prefetch_b8
        __builtin_prefetch(A + (size_t)(bm + srow) * K + kk + 32 + shalf, 0, 1);
        __builtin_prefetch(W + (size_t)(bn + srow) * K + kk + 32 + shalf, 0, 1);
      }
      *(u32x4*)&At[srow * 32 + shalf]     = a0;
      *(u32x4*)&At[srow * 32 + shalf + 8] = a1;
      *(u32x4*)&Bt[srow * 32 + shalf]     = b0;
      *(u32x4*)&Bt[srow * 32 + shalf + 8] = b1;
    }
    __syncthreads();

    // A fragments: lane holds row (laneN), K = base..base+7 and base+16..+23
    v16bf af[2];
#pragma unroll
    for (int i = 0; i < 2; ++i) {
      int r  = wm + i * 16 + laneN;
      int kb = hiHalf * 8;
      u32x4 lo = *(const u32x4*)&At[r * 32 + kb];
      u32x4 hi = *(const u32x4*)&At[r * 32 + kb + 16];
      af[i] = make_frag(lo, hi);
    }
    // B fragments: lane holds column n = W row n, K = base..base+7 and base+16..+23
#pragma unroll
    for (int j = 0; j < 4; ++j) {
      int n  = wn + j * 16 + laneN;
      int kb = hiHalf * 8;
      u32x4 lo = *(const u32x4*)&Bt[n * 32 + kb];
      u32x4 hi = *(const u32x4*)&Bt[n * 32 + kb + 16];
      v16bf bf = make_frag(lo, hi);
#pragma unroll
      for (int i = 0; i < 2; ++i) acc[i][j] = wmma_bf16(af[i], bf, acc[i][j]);
    }
  }

  // Epilogue. C layout: VGPR g -> row g + 8*hiHalf, col = laneN.
#pragma unroll
  for (int i = 0; i < 2; ++i) {
#pragma unroll
    for (int j = 0; j < 4; ++j) {
#pragma unroll
      for (int g = 0; g < 8; ++g) {
        int m = bm + wm + i * 16 + g + hiHalf * 8;
        int n = bn + wn + j * 16 + laneN;
        float v = acc[i][j][g] + bias[n];
        if (MODE == 0) {
          ((float*)out)[(size_t)m * N + n] = v;
        } else if (MODE == 1) {
          int b = m / Srows, s = m % Srows;
          int h = n >> 6, d = n & 63;
          ((unsigned short*)out)[(((size_t)(b * 16 + h) * Srows + s) << 6) + d] =
              f2bf(v);
        } else {
          int b = m / Srows, s = m % Srows;
          int h = n >> 6, d = n & 63;
          ((unsigned short*)out)[((size_t)((b * 16 + h) * 64 + d)) * Srows + s] =
              f2bf(v);
        }
      }
    }
  }
}

// ---------------------------------------------------------------------------
// Flash attention: grid = (S/128, B*H), 256 thr (8 waves).
// K/V tiles staged into LDS by the Tensor Data Mover (wave 0 issues, waits on
// TENSORcnt, workgroup barrier releases the rest). Per 128-key step:
// 16 score WMMAs + online softmax + 16 PV WMMAs.
// qh:[B,H,S,64] kh:[B,H,S,64] vth:[B,H,64,S] (bf16); ctx bf16 [B,S,1024].
// ---------------------------------------------------------------------------
__global__ __launch_bounds__(256) void attn_kernel(
    const unsigned short* __restrict__ qh, const unsigned short* __restrict__ kh,
    const unsigned short* __restrict__ vth, unsigned short* __restrict__ ctx,
    int S) {
  __shared__ unsigned short Kt[128 * 64];      // 128 keys x 64 dh
  __shared__ unsigned short Vt[64 * 128];      // 64 dh  x 128 keys (V^T)
  __shared__ unsigned short Pt[8 * 16 * 128];  // per-wave P staging

  const int tid    = threadIdx.x;
  const int lane   = tid & 31;
  const int w      = tid >> 5;
  const int laneN  = lane & 15;
  const int hiHalf = lane >> 4;
  const int bh     = blockIdx.y;               // b*16 + h
  const int b      = bh >> 4;
  const int h      = bh & 15;
  const int qt     = blockIdx.x;

  const size_t headQ  = (size_t)bh * S * 64;
  const size_t headVT = (size_t)bh * 64 * S;

  const unsigned int ldsK = (unsigned int)(size_t)(void*)Kt;
  const unsigned int ldsV = (unsigned int)(size_t)(void*)Vt;

  // Q A-fragments kept resident (per-wave 16 rows x 64 dh -> 2 K-chunks)
  v16bf aq[2];
  {
    int qrow = qt * 128 + w * 16 + laneN;
    const unsigned short* qr = qh + headQ + (size_t)qrow * 64;
#pragma unroll
    for (int kb = 0; kb < 2; ++kb) {
      int base = kb * 32 + hiHalf * 8;
      u32x4 lo = *(const u32x4*)(qr + base);
      u32x4 hi = *(const u32x4*)(qr + base + 16);
      aq[kb] = make_frag(lo, hi);
    }
  }

  v8f cacc[4];
#pragma unroll
  for (int j = 0; j < 4; ++j) cacc[j] = 0.0f;
  float rowmax[8], rowsum[8];
#pragma unroll
  for (int g = 0; g < 8; ++g) { rowmax[g] = -3.0e38f; rowsum[g] = 0.0f; }

  const float SC    = 0.125f;                 // 1/sqrt(64)
  const float LOG2E = 1.4426950408889634f;
  const float K1    = SC * LOG2E;

  const int nkt = S / 128;
  for (int kt = 0; kt < nkt; ++kt) {
    __syncthreads();  // all waves finished reading Kt/Vt from previous step
    if (w == 0) {
      // K tile: 128 rows x 64 el, row pitch 64  -> Kt[128][64]
      tdm_load_2d_bf16(ldsK, kh + headQ + (size_t)(kt * 128) * 64,
                       64u, 128u, 64u, (unsigned)S, 64u);
      // V^T tile: 64 rows x 128 el, row pitch S -> Vt[64][128]
      tdm_load_2d_bf16(ldsV, vth + headVT + (size_t)kt * 128,
                       128u, 64u, (unsigned)S, 64u, (unsigned)S);
      __builtin_amdgcn_s_wait_tensorcnt(0);
    }
    __syncthreads();  // tiles visible to all waves

    // ---- scores: 16 (q) x 128 (k) = 8 tiles, K-dim = 64 -> 2 chunks ----
    v8f sfrag[8];
#pragma unroll
    for (int nt = 0; nt < 8; ++nt) {
      sfrag[nt] = 0.0f;
#pragma unroll
      for (int kb = 0; kb < 2; ++kb) {
        int n    = nt * 16 + laneN;
        int base = kb * 32 + hiHalf * 8;
        u32x4 lo = *(const u32x4*)&Kt[n * 64 + base];
        u32x4 hi = *(const u32x4*)&Kt[n * 64 + base + 16];
        sfrag[nt] = wmma_bf16(aq[kb], make_frag(lo, hi), sfrag[nt]);
      }
    }

    // ---- online softmax (row stats per VGPR slot, reduced across 16 lanes) --
    float lmax[8];
#pragma unroll
    for (int g = 0; g < 8; ++g) {
      float m = sfrag[0][g];
#pragma unroll
      for (int nt = 1; nt < 8; ++nt) m = fmaxf(m, sfrag[nt][g]);
#pragma unroll
      for (int d = 1; d < 16; d <<= 1) m = fmaxf(m, __shfl_xor(m, d, 32));
      lmax[g] = m * SC;                       // scaled domain
    }
    float rescale[8];
#pragma unroll
    for (int g = 0; g < 8; ++g) {
      float mn   = fmaxf(rowmax[g], lmax[g]);
      rescale[g] = __builtin_amdgcn_exp2f((rowmax[g] - mn) * LOG2E);
      rowmax[g]  = mn;
    }

    unsigned short* pw = &Pt[w * 2048];
    float psum[8];
#pragma unroll
    for (int g = 0; g < 8; ++g) psum[g] = 0.0f;
#pragma unroll
    for (int nt = 0; nt < 8; ++nt) {
#pragma unroll
      for (int g = 0; g < 8; ++g) {
        float p = __builtin_amdgcn_exp2f(sfrag[nt][g] * K1 - rowmax[g] * LOG2E);
        psum[g] += p;
        pw[(g + hiHalf * 8) * 128 + nt * 16 + laneN] = f2bf(p);
      }
    }
#pragma unroll
    for (int g = 0; g < 8; ++g) {
#pragma unroll
      for (int d = 1; d < 16; d <<= 1) psum[g] += __shfl_xor(psum[g], d, 32);
      rowsum[g] = rowsum[g] * rescale[g] + psum[g];
    }
#pragma unroll
    for (int j = 0; j < 4; ++j)
#pragma unroll
      for (int g = 0; g < 8; ++g) cacc[j][g] *= rescale[g];

    __syncthreads();  // P visible (per-wave region; cheap safety barrier)

    // ---- ctx += P (16x128) @ V (128x64): 4 out tiles x 4 K-chunks ----
    v16bf pa[4];
#pragma unroll
    for (int kc = 0; kc < 4; ++kc) {
      int base = kc * 32 + hiHalf * 8;
      u32x4 lo = *(const u32x4*)&pw[laneN * 128 + base];
      u32x4 hi = *(const u32x4*)&pw[laneN * 128 + base + 16];
      pa[kc] = make_frag(lo, hi);
    }
#pragma unroll
    for (int j = 0; j < 4; ++j) {
#pragma unroll
      for (int kc = 0; kc < 4; ++kc) {
        int d    = j * 16 + laneN;
        int base = kc * 32 + hiHalf * 8;
        u32x4 lo = *(const u32x4*)&Vt[d * 128 + base];
        u32x4 hi = *(const u32x4*)&Vt[d * 128 + base + 16];
        cacc[j] = wmma_bf16(pa[kc], make_frag(lo, hi), cacc[j]);
      }
    }
  }

  // ---- normalize and store ctx (bf16, [B,S,1024] with head offset) ----
  const size_t outbase = (size_t)b * S * 1024 + (size_t)h * 64;
#pragma unroll
  for (int g = 0; g < 8; ++g) {
    int   m    = qt * 128 + w * 16 + g + hiHalf * 8;
    float invs = 1.0f / rowsum[g];
#pragma unroll
    for (int j = 0; j < 4; ++j) {
      ctx[outbase + (size_t)m * 1024 + j * 16 + laneN] = f2bf(cacc[j][g] * invs);
    }
  }
}

// ---------------------------------------------------------------------------
// Orchestration
// ---------------------------------------------------------------------------
extern "C" void kernel_launch(void* const* d_in, const int* in_sizes, int n_in,
                              void* d_out, int out_size, void* d_ws,
                              size_t ws_size, hipStream_t stream) {
  const float* Q  = (const float*)d_in[0];
  const float* K  = (const float*)d_in[1];
  const float* V  = (const float*)d_in[2];
  const float* Wq = (const float*)d_in[3];
  const float* bq = (const float*)d_in[4];
  const float* Wk = (const float*)d_in[5];
  const float* bk = (const float*)d_in[6];
  const float* Wv = (const float*)d_in[7];
  const float* bv = (const float*)d_in[8];
  const float* Wo = (const float*)d_in[9];
  const float* bo = (const float*)d_in[10];

  const int B = 2, S = 2048, DK = 1024, M = B * S;
  char* ws = (char*)d_ws;
  const size_t MB = 1ull << 20;
  unsigned short* Qb   = (unsigned short*)(ws + 0 * MB);
  unsigned short* Kb   = (unsigned short*)(ws + 8 * MB);
  unsigned short* Vb   = (unsigned short*)(ws + 16 * MB);
  unsigned short* Wqb  = (unsigned short*)(ws + 24 * MB);
  unsigned short* Wkb  = (unsigned short*)(ws + 26 * MB);
  unsigned short* Wvb  = (unsigned short*)(ws + 28 * MB);
  unsigned short* Wob  = (unsigned short*)(ws + 30 * MB);
  unsigned short* qhb  = (unsigned short*)(ws + 32 * MB);
  unsigned short* khb  = (unsigned short*)(ws + 40 * MB);
  unsigned short* vtb  = (unsigned short*)(ws + 48 * MB);
  unsigned short* ctxb = (unsigned short*)(ws + 56 * MB);

  dim3 blk(256);
  const int nAct = B * S * DK;   // 4194304
  const int nW   = DK * DK;      // 1048576
  cvt_f32_bf16<<<nAct / 1024, blk, 0, stream>>>(Q, Qb, nAct / 4);
  cvt_f32_bf16<<<nAct / 1024, blk, 0, stream>>>(K, Kb, nAct / 4);
  cvt_f32_bf16<<<nAct / 1024, blk, 0, stream>>>(V, Vb, nAct / 4);
  cvt_f32_bf16<<<nW / 1024, blk, 0, stream>>>(Wq, Wqb, nW / 4);
  cvt_f32_bf16<<<nW / 1024, blk, 0, stream>>>(Wk, Wkb, nW / 4);
  cvt_f32_bf16<<<nW / 1024, blk, 0, stream>>>(Wv, Wvb, nW / 4);
  cvt_f32_bf16<<<nW / 1024, blk, 0, stream>>>(Wo, Wob, nW / 4);

  dim3 gg(DK / 128, M / 128);  // (8, 32)
  gemm_bf16<1><<<gg, blk, 0, stream>>>(Qb, Wqb, bq, qhb, M, DK, DK, S);
  gemm_bf16<1><<<gg, blk, 0, stream>>>(Kb, Wkb, bk, khb, M, DK, DK, S);
  gemm_bf16<2><<<gg, blk, 0, stream>>>(Vb, Wvb, bv, vtb, M, DK, DK, S);

  dim3 ga(S / 128, B * 16);    // (16, 32)
  attn_kernel<<<ga, blk, 0, stream>>>(qhb, khb, vtb, ctxb, S);

  gemm_bf16<0><<<gg, blk, 0, stream>>>(ctxb, Wob, bo, d_out, M, DK, DK, S);
}
